// VN_DGCNN_PSEG_mac_63007170232718
// MI455X (gfx1250) — matrix operand order repaired and checked by reading, not compiled, not run on hardware
//
#include <hip/hip_runtime.h>

// ---------------------------------------------------------------------------
// VN-DGCNN part-seg forward for MI455X (gfx1250, wave32).
// Big channel GEMMs -> v_wmma_f32_16x16x32_f16 (fp32 accumulate), 16x64/wave.
// B tile staged through LDS: coalesced b128 global loads -> f16 transposed
// tile -> 16B ds_loads per fragment.  A loads scalar (L0-shared across waves).
// kNN + small vector-neuron convs -> fp32 VALU (index/precision sensitive).
// Train-mode BN handled with two-pass (stats reduce, then normalize) kernels.
// ---------------------------------------------------------------------------

typedef _Float16 v16h __attribute__((ext_vector_type(16)));
typedef _Float16 h8   __attribute__((ext_vector_type(8)));
typedef _Float16 h4   __attribute__((ext_vector_type(4)));
typedef float v8f __attribute__((ext_vector_type(8)));

static constexpr int   Bz  = 4;
static constexpr int   Npt = 2048;
static constexpr int   Kn  = 40;
static constexpr int   Cc  = 21;
static constexpr long  NKL = (long)Npt * Kn;     // 81920
static constexpr int   NKi = (int)NKL;
static constexpr int   LPC = 40;                 // LDS f16 col stride (pad)

#define EPSF   1e-6f
#define BNEPS  1e-5f
#define SLOPE  0.2f
#define NEGINF -3.4e38f

// ------------------------------- utility -----------------------------------
__global__ void k_zero(float* __restrict__ p, int n) {
  int t = blockIdx.x * blockDim.x + threadIdx.x;
  if (t < n) p[t] = 0.f;
}

// ----------------------------- kNN graph -----------------------------------
__global__ void k_xx(const float* __restrict__ xf, long bS, int Cf,
                     float* __restrict__ xx) {
  int t = blockIdx.x * blockDim.x + threadIdx.x;
  if (t >= Bz * Npt) return;
  int b = t / Npt, n = t % Npt;
  const float* xb = xf + (long)b * bS;
  float a = 0.f;
  for (int c = 0; c < Cf; ++c) { float v = xb[c * Npt + n]; a = fmaf(v, v, a); }
  xx[t] = a;
}

// One wave per row: full 2048-wide distance row in LDS, then 40x wave argmax.
__global__ void k_knn(const float* __restrict__ xf, long bS, int Cf,
                      const float* __restrict__ xx, int* __restrict__ idx) {
  __shared__ float sd[4][Npt];
  __shared__ float srow[4][64];
  int wv = threadIdx.x >> 5, lane = threadIdx.x & 31;
  int row = blockIdx.x * 4 + wv;
  int b = row / Npt, n = row % Npt;
  const float* xb = xf + (long)b * bS;
  for (int c = lane; c < Cf; c += 32) srow[wv][c] = xb[c * Npt + n];
  __syncthreads();
  float xn = xx[row];
  for (int m = lane; m < Npt; m += 32) {
    float dot = 0.f;
    for (int c = 0; c < Cf; ++c) dot = fmaf(srow[wv][c], xb[c * Npt + m], dot);
    sd[wv][m] = 2.f * dot - xn - xx[b * Npt + m];
  }
  __syncthreads();
  for (int t = 0; t < Kn; ++t) {
    float best = NEGINF; int bi = 0;
    for (int m = lane; m < Npt; m += 32) {
      float v = sd[wv][m];
      if (v > best) { best = v; bi = m; }
    }
    for (int off = 16; off > 0; off >>= 1) {
      float ov = __shfl_xor(best, off);
      int   oi = __shfl_xor(bi, off);
      if (ov > best || (ov == best && oi < bi)) { best = ov; bi = oi; }
    }
    if (lane == 0) { idx[(long)row * Kn + t] = bi; sd[wv][bi] = NEGINF; }
    __syncthreads();
  }
}

// -------------------- fused graph-feature gather + lin ----------------------
__global__ void k_gather_lin(const float* __restrict__ xin, long bS, int C,
                             const int* __restrict__ idx,
                             const float* __restrict__ W,
                             float* __restrict__ out) {
  long t = (long)blockIdx.x * blockDim.x + threadIdx.x;
  long total = (long)Bz * Npt * Kn;
  if (t >= total) return;
  int k = (int)(t % Kn);
  long nn = t / Kn;
  int n = (int)(nn % Npt);
  int b = (int)(nn / Npt);
  int j = idx[nn * Kn + k];
  const float* xb = xin + (long)b * bS;
  float acc[Cc * 3];
#pragma unroll
  for (int u = 0; u < Cc * 3; ++u) acc[u] = 0.f;
  int CI = 2 * C;
  for (int i = 0; i < CI; ++i) {
    int ii = (i < C) ? i : i - C;
    float c0 = xb[(ii * 3 + 0) * Npt + n];
    float c1 = xb[(ii * 3 + 1) * Npt + n];
    float c2 = xb[(ii * 3 + 2) * Npt + n];
    float f0, f1, f2;
    if (i < C) {
      f0 = xb[(ii * 3 + 0) * Npt + j] - c0;
      f1 = xb[(ii * 3 + 1) * Npt + j] - c1;
      f2 = xb[(ii * 3 + 2) * Npt + j] - c2;
    } else { f0 = c0; f1 = c1; f2 = c2; }
    for (int o = 0; o < Cc; ++o) {
      float w = W[o * CI + i];
      acc[o * 3 + 0] = fmaf(w, f0, acc[o * 3 + 0]);
      acc[o * 3 + 1] = fmaf(w, f1, acc[o * 3 + 1]);
      acc[o * 3 + 2] = fmaf(w, f2, acc[o * 3 + 2]);
    }
  }
  int p = n * Kn + k;
  float* ob = out + (long)b * Cc * 3 * NKL;
  for (int o = 0; o < Cc; ++o)
    for (int s = 0; s < 3; ++s)
      ob[(unsigned)((o * 3 + s) * NKi + p)] = acc[o * 3 + s];
}

// ----------------------------- WMMA GEMM ------------------------------------
// Y[b] (OxS) = W (OxI) * X[b] (IxS); fp32 memory, f16 WMMA, fp32 accumulate.
// Block: 128 threads / 4 waves; each wave owns a 16x64 strip of one 16x256
// block tile.  Main loop stages the 32x256 B tile in LDS (transposed f16),
// then each wave does 4 WMMA per K-step.  Ragged-K tail goes direct-to-global
// with value selects.  All guards are branch-free clamps.
__global__ void __launch_bounds__(128)
k_gemm(const float* __restrict__ W, const float* __restrict__ X,
       float* __restrict__ Y, int O, int I, int S, long xBS, long yBS) {
  __shared__ _Float16 bt[256 * LPC];               // 20 KB f16 tile, [col][row]
  const int t    = threadIdx.x;
  const int lane = t & 31;
  const int wv   = t >> 5;
  const int r    = lane & 15;
  const int hi   = lane >> 4;
  const int m0   = blockIdx.x * 16;
  const int sB   = blockIdx.y * 256;
  const int s0   = sB + wv * 64;
  const int b    = blockIdx.z;
  const float* __restrict__ Xb = X + (long)b * xBS;
  float*       __restrict__ Yb = Y + (long)b * yBS;
  const int mrow  = m0 + r;
  const int mrowC = mrow < O ? mrow : O - 1;       // clamp: harmless rows
  const unsigned aRow = (unsigned)mrowC * (unsigned)I;
  const unsigned aK   = (unsigned)(8 * hi);
  const unsigned uS   = (unsigned)S;
  const int cg  = (t & 63) * 4;                    // local col group (0..252)
  const int rq0 = (t >> 6) * 4;                    // starting row quad
  int gc = sB + cg;                                // clamped global col base
  if (gc + 4 > S) gc = (S >= 4) ? (S - 4) : 0;

  v8f acc[4];
#pragma unroll
  for (int u = 0; u < 4; ++u) acc[u] = (v8f){0.f,0.f,0.f,0.f,0.f,0.f,0.f,0.f};

  int k0 = 0;
  for (; k0 + 32 <= I; k0 += 32) {
    __syncthreads();                               // tile consumed
#pragma unroll
    for (int rq = 0; rq < 32; rq += 8) {           // stage 32x256 tile
      int row = rq + rq0;
      float4 v0 = *(const float4*)(Xb + (unsigned)(k0 + row + 0) * uS + (unsigned)gc);
      float4 v1 = *(const float4*)(Xb + (unsigned)(k0 + row + 1) * uS + (unsigned)gc);
      float4 v2 = *(const float4*)(Xb + (unsigned)(k0 + row + 2) * uS + (unsigned)gc);
      float4 v3 = *(const float4*)(Xb + (unsigned)(k0 + row + 3) * uS + (unsigned)gc);
      *(h4*)(&bt[(cg + 0) * LPC + row]) =
          (h4){(_Float16)v0.x, (_Float16)v1.x, (_Float16)v2.x, (_Float16)v3.x};
      *(h4*)(&bt[(cg + 1) * LPC + row]) =
          (h4){(_Float16)v0.y, (_Float16)v1.y, (_Float16)v2.y, (_Float16)v3.y};
      *(h4*)(&bt[(cg + 2) * LPC + row]) =
          (h4){(_Float16)v0.z, (_Float16)v1.z, (_Float16)v2.z, (_Float16)v3.z};
      *(h4*)(&bt[(cg + 3) * LPC + row]) =
          (h4){(_Float16)v0.w, (_Float16)v1.w, (_Float16)v2.w, (_Float16)v3.w};
    }
    if (k0 + 64 <= I)                              // prefetch next A block
      __builtin_prefetch(&W[aRow + (unsigned)(k0 + 32) + aK], 0, 3);
    __syncthreads();                               // tile ready
    v16h a;
#pragma unroll
    for (int e = 0; e < 16; ++e) {                 // A 16x32 f16 layout
      unsigned ka = (unsigned)k0 + (unsigned)((e & 7) + 16 * (e >> 3)) + aK;
      a[e] = (_Float16)W[aRow + ka];
    }
#pragma unroll
    for (int u = 0; u < 4; ++u) {
      int colL = wv * 64 + u * 16 + r;
      const _Float16* fp = &bt[colL * LPC + 8 * hi];
      h8 blo = *(const h8*)(fp);
      h8 bhi = *(const h8*)(fp + 16);
      v16h bm;
#pragma unroll
      for (int e = 0; e < 8; ++e) { bm[e] = blo[e]; bm[8 + e] = bhi[e]; }
      acc[u] = __builtin_amdgcn_wmma_f32_16x16x32_f16(false, a, false, bm,
                                                      (short)0, acc[u], false, false);
    }
  }
  if (k0 < I) {                                    // ragged-K tail (once)
    unsigned colC[4];
#pragma unroll
    for (int u = 0; u < 4; ++u) {
      int c = s0 + 16 * u + r;
      colC[u] = (unsigned)(c < S ? c : S - 1);
    }
    v16h a;
#pragma unroll
    for (int e = 0; e < 16; ++e) {
      int ka = k0 + (e & 7) + 8 * hi + 16 * (e >> 3);
      int kc = ka < I ? ka : I - 1;
      float v = W[aRow + (unsigned)kc];
      a[e] = ka < I ? (_Float16)v : (_Float16)0.f;  // select, not branch
    }
#pragma unroll
    for (int u = 0; u < 4; ++u) {
      v16h bm;
#pragma unroll
      for (int e = 0; e < 16; ++e) {
        int kb = k0 + (e & 7) + 8 * hi + 16 * (e >> 3);
        int kc = kb < I ? kb : I - 1;
        float v = Xb[(unsigned)kc * uS + colC[u]];
        bm[e] = kb < I ? (_Float16)v : (_Float16)0.f;
      }
      acc[u] = __builtin_amdgcn_wmma_f32_16x16x32_f16(false, a, false, bm,
                                                      (short)0, acc[u], false, false);
    }
  }
#pragma unroll
  for (int u = 0; u < 4; ++u) {
    int col = s0 + 16 * u + r;
    if (col < S) {
#pragma unroll
      for (int v = 0; v < 8; ++v) {
        int row = m0 + v + 8 * hi;                 // C/D 16x16 layout
        if (row < O) Yb[(unsigned)row * uS + (unsigned)col] = acc[u][v];
      }
    }
  }
}

// ----------------------- vector-neuron BN + LeakyLLR ------------------------
__global__ void k_vnstats(const float* __restrict__ q, long bS, long SP,
                          float* __restrict__ stats) {
  int o = blockIdx.x;
  __shared__ float s1s[256], s2s[256];
  float s1 = 0.f, s2 = 0.f;
  long per = (long)Bz * SP;
  for (long p = (long)blockIdx.y * blockDim.x + threadIdx.x; p < per;
       p += (long)gridDim.y * blockDim.x) {
    int  b  = (int)(p / SP);
    long pp = p % SP;
    const float* base = q + (long)b * bS + (long)o * 3 * SP + pp;
    float q0 = base[0], q1 = base[SP], q2 = base[2 * SP];
    float nr = sqrtf(q0 * q0 + q1 * q1 + q2 * q2) + EPSF;
    s1 += nr; s2 += nr * nr;
  }
  s1s[threadIdx.x] = s1; s2s[threadIdx.x] = s2;
  __syncthreads();
  for (int w = 128; w > 0; w >>= 1) {
    if ((int)threadIdx.x < w) {
      s1s[threadIdx.x] += s1s[threadIdx.x + w];
      s2s[threadIdx.x] += s2s[threadIdx.x + w];
    }
    __syncthreads();
  }
  if (threadIdx.x == 0) {
    atomicAdd(&stats[2 * o + 0], s1s[0]);
    atomicAdd(&stats[2 * o + 1], s2s[0]);
  }
}

__global__ void k_llr_finish(float* __restrict__ q, long qBS,
                             const float* __restrict__ d, long dBS, int dShared,
                             const float* __restrict__ stats,
                             const float* __restrict__ gamma,
                             const float* __restrict__ beta,
                             int CO, long SP, float invCnt) {
  long t = (long)blockIdx.x * blockDim.x + threadIdx.x;
  long total = (long)Bz * CO * SP;
  if (t >= total) return;
  long p = t % SP;
  long rr = t / SP;
  int o = (int)(rr % CO);
  int b = (int)(rr / CO);
  float* qp = q + (long)b * qBS + (long)o * 3 * SP + p;
  const float* dp = d + (long)b * dBS + (long)(dShared ? 0 : o) * 3 * SP + p;
  float q0 = qp[0], q1 = qp[SP], q2 = qp[2 * SP];
  float d0 = dp[0], d1 = dp[SP], d2 = dp[2 * SP];
  float nr = sqrtf(q0 * q0 + q1 * q1 + q2 * q2) + EPSF;
  float mean = stats[2 * o] * invCnt;
  float var  = stats[2 * o + 1] * invCnt - mean * mean;
  float bnv  = (nr - mean) * rsqrtf(var + BNEPS) * gamma[o] + beta[o];
  float pf = bnv / nr;
  float p0 = q0 * pf, p1 = q1 * pf, p2 = q2 * pf;
  float dot = p0 * d0 + p1 * d1 + p2 * d2;
  float dsq = d0 * d0 + d1 * d1 + d2 * d2;
  float rfc = dot / (dsq + EPSF);
  float o0 = p0, o1 = p1, o2 = p2;
  if (dot < 0.f) { o0 = p0 - rfc * d0; o1 = p1 - rfc * d1; o2 = p2 - rfc * d2; }
  qp[0]      = SLOPE * p0 + (1.f - SLOPE) * o0;
  qp[SP]     = SLOPE * p1 + (1.f - SLOPE) * o1;
  qp[2 * SP] = SLOPE * p2 + (1.f - SLOPE) * o2;
}

// ------------------------------ vn_max_pool ---------------------------------
__global__ void k_pool(const float* __restrict__ h, const float* __restrict__ Wp,
                       float* __restrict__ out, int outCT, int outCOff) {
  long t = (long)blockIdx.x * blockDim.x + threadIdx.x;
  long total = (long)Bz * Cc * Npt;
  if (t >= total) return;
  int n = (int)(t % Npt);
  int o = (int)((t / Npt) % Cc);
  int b = (int)(t / ((long)Npt * Cc));
  const float* hb = h + (long)b * Cc * 3 * NKL;
  int pb = n * Kn;
  float best = NEGINF; int kb = 0;
  for (int k = 0; k < Kn; ++k) {
    float d0 = 0.f, d1 = 0.f, d2 = 0.f;
    for (int i = 0; i < Cc; ++i) {
      float w = Wp[o * Cc + i];
      unsigned hp = (unsigned)(i * 3 * NKi + pb + k);
      d0 = fmaf(w, hb[hp], d0);
      d1 = fmaf(w, hb[hp + (unsigned)NKi], d1);
      d2 = fmaf(w, hb[hp + (unsigned)(2 * NKi)], d2);
    }
    unsigned ho = (unsigned)(o * 3 * NKi + pb + k);
    float dot = hb[ho] * d0 + hb[ho + (unsigned)NKi] * d1 +
                hb[ho + (unsigned)(2 * NKi)] * d2;
    if (dot > best) { best = dot; kb = k; }
  }
  unsigned ho = (unsigned)(o * 3 * NKi + pb + kb);
  for (int s = 0; s < 3; ++s)
    out[((long)(b * outCT + outCOff + o) * 3 + s) * Npt + n] =
        hb[ho + (unsigned)(s * NKi)];
}

// ------------------- misc middle-of-network kernels -------------------------
__global__ void k_mean_bcast(float* __restrict__ h) {
  int s = blockIdx.x % 3;
  int c = (blockIdx.x / 3) % 341;
  int b = blockIdx.x / (3 * 341);
  const float* src = h + ((long)(b * 682 + c) * 3 + s) * Npt;
  __shared__ float red[256];
  float a = 0.f;
  for (int n = threadIdx.x; n < Npt; n += 256) a += src[n];
  red[threadIdx.x] = a;
  __syncthreads();
  for (int w = 128; w > 0; w >>= 1) {
    if ((int)threadIdx.x < w) red[threadIdx.x] += red[threadIdx.x + w];
    __syncthreads();
  }
  float m = red[0] * (1.f / Npt);
  float* dst = h + ((long)(b * 682 + 341 + c) * 3 + s) * Npt;
  for (int n = threadIdx.x; n < Npt; n += 256) dst[n] = m;
}

__global__ void k_rot(float* __restrict__ h, const float* __restrict__ z0, int C) {
  long t = (long)blockIdx.x * blockDim.x + threadIdx.x;
  long total = (long)Bz * C * Npt;
  if (t >= total) return;
  int n = (int)(t % Npt);
  int c = (int)((t / Npt) % C);
  int b = (int)(t / ((long)Npt * C));
  float* hp = h + ((long)(b * C + c) * 3) * Npt + n;
  float h0 = hp[0], h1 = hp[Npt], h2 = hp[2 * Npt];
  const float* zb = z0 + (long)b * 9 * Npt + n;
  float r0 = h0 * zb[0 * Npt] + h1 * zb[1 * Npt] + h2 * zb[2 * Npt];
  float r1 = h0 * zb[3 * Npt] + h1 * zb[4 * Npt] + h2 * zb[5 * Npt];
  float r2 = h0 * zb[6 * Npt] + h1 * zb[7 * Npt] + h2 * zb[8 * Npt];
  hp[0] = r0; hp[Npt] = r1; hp[2 * Npt] = r2;
}

__global__ void k_gmax(const float* __restrict__ h, float* __restrict__ g) {
  long base = (long)blockIdx.x * Npt;
  __shared__ float red[256];
  float a = NEGINF;
  for (int n = threadIdx.x; n < Npt; n += 256) a = fmaxf(a, h[base + n]);
  red[threadIdx.x] = a;
  __syncthreads();
  for (int w = 128; w > 0; w >>= 1) {
    if ((int)threadIdx.x < w)
      red[threadIdx.x] = fmaxf(red[threadIdx.x], red[threadIdx.x + w]);
    __syncthreads();
  }
  if (threadIdx.x == 0) g[blockIdx.x] = red[0];
}

__global__ void k_conv7(const float* __restrict__ l, const float* __restrict__ W,
                        const float* __restrict__ gm, const float* __restrict__ bt,
                        float* __restrict__ lv) {
  int o = threadIdx.x;
  float y[Bz];
  float s1 = 0.f, s2 = 0.f;
  for (int b = 0; b < Bz; ++b) {
    float a = 0.f;
    for (int i = 0; i < 16; ++i) a = fmaf(W[o * 16 + i], l[b * 16 + i], a);
    y[b] = a; s1 += a; s2 += a * a;
  }
  float mean = s1 * (1.f / Bz);
  float var  = s2 * (1.f / Bz) - mean * mean;
  float sc   = rsqrtf(var + BNEPS) * gm[o];
  for (int b = 0; b < Bz; ++b) {
    float z = (y[b] - mean) * sc + bt[o];
    lv[b * 64 + o] = z >= 0.f ? z : SLOPE * z;
  }
}

__global__ void k_build_hin(const float* __restrict__ g, const float* __restrict__ lv,
                            const float* __restrict__ x123, float* __restrict__ hin) {
  long t = (long)blockIdx.x * blockDim.x + threadIdx.x;
  long total = (long)Bz * 2299 * Npt;
  if (t >= total) return;
  int n = (int)(t % Npt);
  int c = (int)((t / Npt) % 2299);
  int b = (int)(t / ((long)Npt * 2299));
  float v;
  if (c < 2046)      v = g[b * 2046 + c];
  else if (c < 2110) v = lv[b * 64 + (c - 2046)];
  else               v = x123[(long)b * 189 * Npt + (c - 2110) * Npt + n];
  hin[t] = v;
}

__global__ void k_bnstats(const float* __restrict__ y, int CO,
                          float* __restrict__ stats) {
  int o = blockIdx.x;
  __shared__ float s1s[256], s2s[256];
  float s1 = 0.f, s2 = 0.f;
  long per = (long)Bz * Npt;
  for (long p = (long)blockIdx.y * blockDim.x + threadIdx.x; p < per;
       p += (long)gridDim.y * blockDim.x) {
    int b = (int)(p / Npt);
    int n = (int)(p % Npt);
    float v = y[((long)(b * CO + o)) * Npt + n];
    s1 += v; s2 += v * v;
  }
  s1s[threadIdx.x] = s1; s2s[threadIdx.x] = s2;
  __syncthreads();
  for (int w = 128; w > 0; w >>= 1) {
    if ((int)threadIdx.x < w) {
      s1s[threadIdx.x] += s1s[threadIdx.x + w];
      s2s[threadIdx.x] += s2s[threadIdx.x + w];
    }
    __syncthreads();
  }
  if (threadIdx.x == 0) {
    atomicAdd(&stats[2 * o + 0], s1s[0]);
    atomicAdd(&stats[2 * o + 1], s2s[0]);
  }
}

__global__ void k_bn_lrelu(float* __restrict__ y, int CO,
                           const float* __restrict__ stats,
                           const float* __restrict__ gm,
                           const float* __restrict__ bt, float invCnt) {
  long t = (long)blockIdx.x * blockDim.x + threadIdx.x;
  long total = (long)Bz * CO * Npt;
  if (t >= total) return;
  int o = (int)((t / Npt) % CO);
  float mean = stats[2 * o] * invCnt;
  float var  = stats[2 * o + 1] * invCnt - mean * mean;
  float z = (y[t] - mean) * rsqrtf(var + BNEPS) * gm[o] + bt[o];
  y[t] = z >= 0.f ? z : SLOPE * z;
}

// ------------------------------ host side -----------------------------------
extern "C" void kernel_launch(void* const* d_in, const int* in_sizes, int n_in,
                              void* d_out, int out_size, void* d_ws, size_t ws_size,
                              hipStream_t stream) {
  (void)out_size; (void)ws_size;
  const float* x = (const float*)d_in[0];
  const float* lbl = (const float*)d_in[1];

  // ---- resolve param leaves (insertion order / sorted pytree / flat concat)
  const float *c1W,*c1Wd,*c1g,*c1b, *c2W,*c2Wd,*c2g,*c2b, *c3W,*c3Wd,*c3g,*c3b;
  const float *c4W,*c4Wd,*c4g,*c4b, *c5W,*c5Wd,*c5g,*c5b;
  const float *pool1,*pool2,*pool3;
  const float *c6W,*c6Wd,*c6g,*c6b, *v1W,*v1Wd,*v1g,*v1b, *v2W,*v2Wd,*v2g,*v2b;
  const float *vlin, *c7W,*c7g,*c7b, *c8W,*c8g,*c8b, *c9W,*c9g,*c9b;
  const float *c10W,*c10g,*c10b, *c11W;

  if (n_in >= 51 && in_sizes[6] == 441) {           // insertion order
    auto P = [&](int i){ return (const float*)d_in[i]; };
    c1W=P(2);c1Wd=P(3);c1g=P(4);c1b=P(5); c2W=P(6);c2Wd=P(7);c2g=P(8);c2b=P(9);
    c3W=P(10);c3Wd=P(11);c3g=P(12);c3b=P(13); c4W=P(14);c4Wd=P(15);c4g=P(16);c4b=P(17);
    c5W=P(18);c5Wd=P(19);c5g=P(20);c5b=P(21);
    pool1=P(22);pool2=P(23);pool3=P(24);
    c6W=P(25);c6Wd=P(26);c6g=P(27);c6b=P(28); v1W=P(29);v1Wd=P(30);v1g=P(31);v1b=P(32);
    v2W=P(33);v2Wd=P(34);v2g=P(35);v2b=P(36); vlin=P(37);
    c7W=P(38);c7g=P(39);c7b=P(40); c8W=P(41);c8g=P(42);c8b=P(43);
    c9W=P(44);c9g=P(45);c9b=P(46); c10W=P(47);c10g=P(48);c10b=P(49); c11W=P(50);
  } else if (n_in >= 51) {                          // sorted pytree key order
    auto P = [&](int i){ return (const float*)d_in[i]; };
    c1W=P(2);c1Wd=P(3);c1g=P(4);c1b=P(5); c10W=P(6);c10g=P(7);c10b=P(8); c11W=P(9);
    c2W=P(10);c2Wd=P(11);c2g=P(12);c2b=P(13); c3W=P(14);c3Wd=P(15);c3g=P(16);c3b=P(17);
    c4W=P(18);c4Wd=P(19);c4g=P(20);c4b=P(21); c5W=P(22);c5Wd=P(23);c5g=P(24);c5b=P(25);
    c6W=P(26);c6Wd=P(27);c6g=P(28);c6b=P(29); c7W=P(30);c7g=P(31);c7b=P(32);
    c8W=P(33);c8g=P(34);c8b=P(35); c9W=P(36);c9g=P(37);c9b=P(38);
    pool1=P(39);pool2=P(40);pool3=P(41);
    v1W=P(42);v1Wd=P(43);v1g=P(44);v1b=P(45); v2W=P(46);v2Wd=P(47);v2g=P(48);v2b=P(49);
    vlin=P(50);
  } else {                                          // single concatenated buffer
    const float* base = (const float*)d_in[2];
    long o = 0;
    auto P = [&](long nEl){ const float* p = base + o; o += nEl; return p; };
    c1W=P(42);c1Wd=P(42);c1g=P(21);c1b=P(21); c2W=P(441);c2Wd=P(441);c2g=P(21);c2b=P(21);
    c3W=P(882);c3Wd=P(882);c3g=P(21);c3b=P(21); c4W=P(441);c4Wd=P(441);c4g=P(21);c4b=P(21);
    c5W=P(882);c5Wd=P(882);c5g=P(21);c5b=P(21);
    pool1=P(441);pool2=P(441);pool3=P(441);
    c6W=P(21483);c6Wd=P(63);c6g=P(341);c6b=P(341);
    v1W=P(232562);v1Wd=P(232562);v1g=P(341);v1b=P(341);
    v2W=P(57970);v2Wd=P(57970);v2g=P(170);v2b=P(170);
    vlin=P(510);
    c7W=P(1024);c7g=P(64);c7b=P(64); c8W=P(588544);c8g=P(256);c8b=P(256);
    c9W=P(65536);c9g=P(256);c9b=P(256); c10W=P(32768);c10g=P(128);c10b=P(128);
    c11W=P(6400);
  }

  // ---- workspace layout (floats)
  float* ws = (float*)d_ws;
  const long GBsz = 20644096;                // 21*3*N*K*B padded
  float* P0   = ws;
  float* P1   = ws + GBsz;
  float* P2   = ws + 2 * GBsz;
  float* x123 = ws + 3 * GBsz;               // (B,63,3,N) = 1,548,288
  float* h682 = x123 + 1548544;              // (B,682,3,N) = 16,760,832
  float* xxb  = h682 + 16761088;             // 8192
  int*   idxb = (int*)(xxb + 8192);          // 327,680 ints
  float* stats= (float*)(idxb + 327680);     // 2048
  float* d6   = stats + 2048;                // (B,1,3,N) = 24,576
  // reuse inside big buffers:
  float* z1q = P0;  float* z1d = P0 + 8500000;
  float* z2q = P1;  float* z2d = P1 + 4500000;
  float* z0b = P2;  float* gbuf = P2 + 100000; float* lvb = P2 + 110000;
  float* hin = P0;
  float* y8  = P1;  float* y9 = P1 + 2200000; float* y10 = P1 + 4400000;

  const long GQBS = (long)Cc * 3 * NKL;      // graph q/d per-batch stride
  const int  S3NK = (int)(3 * NKL);          // 245760
  const int  S3N  = 3 * Npt;                 // 6144

  auto gemm = [&](const float* W, const float* X, float* Y, int O, int I, int S,
                  long xBS, long yBS){
    dim3 g((O + 15) / 16, (S + 255) / 256, Bz);   // 4 waves x 64 cols per block
    k_gemm<<<g, 128, 0, stream>>>(W, X, Y, O, I, S, xBS, yBS);
  };
  auto vnbn = [&](float* q, long qBS, const float* d, long dBS, int dSh,
                  const float* gm, const float* bt, int CO, long SP){
    k_zero<<<(2 * CO + 255) / 256, 256, 0, stream>>>(stats, 2 * CO);
    dim3 gs(CO, 64);
    k_vnstats<<<gs, 256, 0, stream>>>(q, qBS, SP, stats);
    long total = (long)Bz * CO * SP;
    k_llr_finish<<<(unsigned)((total + 255) / 256), 256, 0, stream>>>(
        q, qBS, d, dBS, dSh, stats, gm, bt, CO, SP, 1.f / (float)(Bz * SP));
  };
  auto bn = [&](float* y, int CO, const float* gm, const float* bt){
    k_zero<<<(2 * CO + 255) / 256, 256, 0, stream>>>(stats, 2 * CO);
    dim3 gs(CO, 32);
    k_bnstats<<<gs, 256, 0, stream>>>(y, CO, stats);
    long total = (long)Bz * CO * Npt;
    k_bn_lrelu<<<(unsigned)((total + 255) / 256), 256, 0, stream>>>(
        y, CO, stats, gm, bt, 1.f / (float)(Bz * Npt));
  };
  const long totG = (long)Bz * Npt * Kn;
  const unsigned gbG = (unsigned)((totG + 255) / 256);
  const unsigned gbPool = (unsigned)(((long)Bz * Cc * Npt + 255) / 256);

  // ---------------- stage 1 (C=1, xf = x (B,3,N)) ----------------
  k_xx <<<(Bz * Npt + 255) / 256, 256, 0, stream>>>(x, 3L * Npt, 3, xxb);
  k_knn<<<Bz * Npt / 4, 128, 0, stream>>>(x, 3L * Npt, 3, xxb, idxb);
  k_gather_lin<<<gbG, 256, 0, stream>>>(x, 3L * Npt, 1, idxb, c1W,  P0);
  k_gather_lin<<<gbG, 256, 0, stream>>>(x, 3L * Npt, 1, idxb, c1Wd, P1);
  vnbn(P0, GQBS, P1, GQBS, 0, c1g, c1b, Cc, NKL);
  gemm(c2W,  P0, P1, Cc, Cc, S3NK, GQBS, GQBS);
  gemm(c2Wd, P0, P2, Cc, Cc, S3NK, GQBS, GQBS);
  vnbn(P1, GQBS, P2, GQBS, 0, c2g, c2b, Cc, NKL);
  k_pool<<<gbPool, 256, 0, stream>>>(P1, pool1, x123, 63, 0);

  // ---------------- stage 2 (C=21, xf = x1 in x123[0:21)) ----------------
  k_xx <<<(Bz * Npt + 255) / 256, 256, 0, stream>>>(x123, 189L * Npt, 63, xxb);
  k_knn<<<Bz * Npt / 4, 128, 0, stream>>>(x123, 189L * Npt, 63, xxb, idxb);
  k_gather_lin<<<gbG, 256, 0, stream>>>(x123, 189L * Npt, Cc, idxb, c3W,  P0);
  k_gather_lin<<<gbG, 256, 0, stream>>>(x123, 189L * Npt, Cc, idxb, c3Wd, P1);
  vnbn(P0, GQBS, P1, GQBS, 0, c3g, c3b, Cc, NKL);
  gemm(c4W,  P0, P1, Cc, Cc, S3NK, GQBS, GQBS);
  gemm(c4Wd, P0, P2, Cc, Cc, S3NK, GQBS, GQBS);
  vnbn(P1, GQBS, P2, GQBS, 0, c4g, c4b, Cc, NKL);
  k_pool<<<gbPool, 256, 0, stream>>>(P1, pool2, x123, 63, 21);

  // ---------------- stage 3 (C=21, xf = x2 in x123[21:42)) ----------------
  const float* x2base = x123 + 63L * Npt;
  k_xx <<<(Bz * Npt + 255) / 256, 256, 0, stream>>>(x2base, 189L * Npt, 63, xxb);
  k_knn<<<Bz * Npt / 4, 128, 0, stream>>>(x2base, 189L * Npt, 63, xxb, idxb);
  k_gather_lin<<<gbG, 256, 0, stream>>>(x2base, 189L * Npt, Cc, idxb, c5W,  P0);
  k_gather_lin<<<gbG, 256, 0, stream>>>(x2base, 189L * Npt, Cc, idxb, c5Wd, P1);
  vnbn(P0, GQBS, P1, GQBS, 0, c5g, c5b, Cc, NKL);
  k_pool<<<gbPool, 256, 0, stream>>>(P0, pool3, x123, 63, 42);

  // ---------------- conv6 (shared Wd) + mean concat ----------------
  gemm(c6W,  x123, h682, 341, 63, S3N, 189L * Npt, 682L * 3 * Npt);
  gemm(c6Wd, x123, d6,     1, 63, S3N, 189L * Npt,   3L * Npt);
  vnbn(h682, 682L * 3 * Npt, d6, 3L * Npt, 1, c6g, c6b, 341, Npt);
  k_mean_bcast<<<Bz * 341 * 3, 256, 0, stream>>>(h682);

  // ---------------- vn1, vn2, rotation ----------------
  gemm(v1W,  h682, z1q, 341, 682, S3N, 682L * 3 * Npt, 341L * 3 * Npt);
  gemm(v1Wd, h682, z1d, 341, 682, S3N, 682L * 3 * Npt, 341L * 3 * Npt);
  vnbn(z1q, 341L * 3 * Npt, z1d, 341L * 3 * Npt, 0, v1g, v1b, 341, Npt);
  gemm(v2W,  z1q, z2q, 170, 341, S3N, 341L * 3 * Npt, 170L * 3 * Npt);
  gemm(v2Wd, z1q, z2d, 170, 341, S3N, 341L * 3 * Npt, 170L * 3 * Npt);
  vnbn(z2q, 170L * 3 * Npt, z2d, 170L * 3 * Npt, 0, v2g, v2b, 170, Npt);
  gemm(vlin, z2q, z0b, 3, 170, S3N, 170L * 3 * Npt, 9L * Npt);
  k_rot<<<(unsigned)(((long)Bz * 682 * Npt + 255) / 256), 256, 0, stream>>>(h682, z0b, 682);
  k_rot<<<(unsigned)(((long)Bz * 63 * Npt + 255) / 256), 256, 0, stream>>>(x123, z0b, 63);

  // ---------------- global max, label branch, head ----------------
  k_gmax<<<Bz * 2046, 256, 0, stream>>>(h682, gbuf);
  k_conv7<<<1, 64, 0, stream>>>(lbl, c7W, c7g, c7b, lvb);
  k_build_hin<<<(unsigned)(((long)Bz * 2299 * Npt + 255) / 256), 256, 0, stream>>>(
      gbuf, lvb, x123, hin);
  gemm(c8W, hin, y8, 256, 2299, Npt, 2299L * Npt, 256L * Npt);
  bn(y8, 256, c8g, c8b);
  gemm(c9W, y8, y9, 256, 256, Npt, 256L * Npt, 256L * Npt);
  bn(y9, 256, c9g, c9b);
  gemm(c10W, y9, y10, 128, 256, Npt, 256L * Npt, 128L * Npt);
  bn(y10, 128, c10g, c10b);
  gemm(c11W, y10, (float*)d_out, 50, 128, Npt, 128L * Npt, 50L * Npt);
}
